// DeformableTransformerDecoderLayer_17755394802028
// MI455X (gfx1250) — matrix operand order repaired and checked by reading, not compiled, not run on hardware
//
#include <hip/hip_runtime.h>
#include <hip/hip_bf16.h>

// ---------------------------------------------------------------------------
// Problem constants (from the reference)
// ---------------------------------------------------------------------------
#define B_    8
#define NQ_   900
#define D_    256
#define NH_   8
#define NL_   4
#define NP_   4
#define DH_   32
#define DFF_  1024
#define LV_   13294          // 100*100 + 50*50 + 25*25 + 13*13
#define NTOK_ (B_ * NQ_)     // 7200
#define MTOK_ (B_ * LV_)     // 106352

typedef __attribute__((ext_vector_type(16))) _Float16 v16h;
typedef __attribute__((ext_vector_type(8)))  float    v8f;

// ---------------------------------------------------------------------------
// WMMA fragment loaders: unconditional wide loads, f32 -> f16 in-register.
// Requirements (met by all call sites): K % 32 == 0, lda/ldw even,
// base pointers 16B-aligned. Edge rows/cols are handled by the CALLER via
// address clamping (loads) + store guards, so no per-element predication.
//
// A-matrix 16x32 f16 layout (ISA 7.12.2): lane = m + 16*hi; element e
// (VGPR e/2, half e&1) -> k = (v<4 ? 2v : 8+2v) + 8*hi + (e&1), v=e>>1.
// Pairs (2v,2v+1) are K-contiguous -> float2 loads.
// ---------------------------------------------------------------------------
__device__ __forceinline__ v16h load_a_frag(const float* __restrict__ A, long row,
                                            int lda, int k0, int hi)
{
    v16h f;
    const float* p = A + row * (long)lda + k0 + 8 * hi;
#pragma unroll
    for (int v = 0; v < 8; ++v) {
        const int kb = (v < 4) ? (2 * v) : (8 + 2 * v);   // 0,2,4,6,16,18,20,22
        float2 x = *(const float2*)(p + kb);
        f[2 * v]     = (_Float16)x.x;
        f[2 * v + 1] = (_Float16)x.y;
    }
    return f;
}

// B-matrix 32x16 f16 (B[k][n] = W[n][k]): lane&15 = n; element e ->
// k = (e<8 ? e : 8+e) + 8*hi (same as A) => two runs of 8 contiguous
// floats -> four float4 loads.
__device__ __forceinline__ v16h load_b_frag(const float* __restrict__ W, long col,
                                            int ldw, int k0, int hi)
{
    v16h f;
    const float* p = W + col * (long)ldw + k0 + 8 * hi;
#pragma unroll
    for (int i = 0; i < 4; ++i) {
        float4 x = *(const float4*)(p + 4 * i + (i >= 2 ? 8 : 0));
        f[4 * i + 0] = (_Float16)x.x;
        f[4 * i + 1] = (_Float16)x.y;
        f[4 * i + 2] = (_Float16)x.z;
        f[4 * i + 3] = (_Float16)x.w;
    }
    return f;
}

__device__ __forceinline__ v8f wmma_f16(v16h a, v16h b, v8f c)
{
    return __builtin_amdgcn_wmma_f32_16x16x32_f16(false, a, false, b,
                                                  (short)0, c, false, false);
}

// ---------------------------------------------------------------------------
// Generic GEMM: C[M,N] = alpha * (A[M,K] @ W[N,K]^T) + bias (+ relu)
// One wave computes a 32x32 output block = 2x2 WMMA tiles (4 wmma / K-step,
// 2x register reuse of each fragment). Requires K % 32 == 0; M,N edges via
// clamp+store-guard.
// ---------------------------------------------------------------------------
__global__ void gemm_wmma_kernel(const float* __restrict__ A, const float* __restrict__ W,
                                 const float* __restrict__ bias, float* __restrict__ C,
                                 int M, int N, int K, int lda, int ldw, int ldc,
                                 float alpha, int relu)
{
    const int lane = threadIdx.x;        // 0..31, wave32
    const int hi   = lane >> 4;
    const int l16  = lane & 15;

    const long mbase = (long)blockIdx.y * 32;
    const long nbase = (long)blockIdx.x * 32;

    // Load-address rows/cols, clamped so edge tiles read valid memory.
    const long rA0 = min(mbase + l16,      (long)M - 1);
    const long rA1 = min(mbase + 16 + l16, (long)M - 1);
    const long cB0 = min(nbase + l16,      (long)N - 1);
    const long cB1 = min(nbase + 16 + l16, (long)N - 1);

    v8f acc00 = {}, acc01 = {}, acc10 = {}, acc11 = {};

    for (int k0 = 0; k0 < K; k0 += 32) {
        v16h a0 = load_a_frag(A, rA0, lda, k0, hi);
        v16h a1 = load_a_frag(A, rA1, lda, k0, hi);
        v16h b0 = load_b_frag(W, cB0, ldw, k0, hi);
        v16h b1 = load_b_frag(W, cB1, ldw, k0, hi);
        acc00 = wmma_f16(a0, b0, acc00);
        acc01 = wmma_f16(a0, b1, acc01);
        acc10 = wmma_f16(a1, b0, acc10);
        acc11 = wmma_f16(a1, b1, acc11);
    }

    // C/D layout: VGPR r, lanes 0-15 -> M = r, lanes 16-31 -> M = r + 8.
#pragma unroll
    for (int tj = 0; tj < 2; ++tj) {
        const long col = nbase + tj * 16 + l16;
        if (col >= N) continue;
        const float bv = bias ? bias[col] : 0.0f;
#pragma unroll
        for (int ti = 0; ti < 2; ++ti) {
            const v8f* accp = ti ? (tj ? &acc11 : &acc10)
                                 : (tj ? &acc01 : &acc00);
#pragma unroll
            for (int r = 0; r < 8; ++r) {
                const long m = mbase + ti * 16 + r + 8 * hi;
                if (m < M) {
                    float x = alpha * (*accp)[r] + bv;
                    if (relu) x = fmaxf(x, 0.0f);
                    C[m * (long)ldc + col] = x;
                }
            }
        }
    }
}

// ---------------------------------------------------------------------------
// Fused self-attention with LDS-tiled K/V (online softmax).
// Grid: (ceil(NQ/128), NH, B); block = 128 threads, one query per thread.
// K/V chunks of 64 keys are staged in LDS (16 KB of the 320 KB WGP pool)
// and shared by all 128 queries -> ~128x less L2 read traffic than the
// one-thread-per-(b,q,h) formulation.
// qkv layout: [B*NQ, 768] = [q | k | v], head h occupies cols h*32..h*32+31.
// ---------------------------------------------------------------------------
#define AQ_ 128   // queries per block
#define AK_ 64    // keys per LDS chunk

__global__ void self_attn_kernel(const float* __restrict__ qkv, float* __restrict__ out)
{
    const int tid = threadIdx.x;              // 0..127
    const int b   = blockIdx.z;
    const int h   = blockIdx.y;
    const int qg  = blockIdx.x * AQ_ + tid;   // global query id (maybe OOB)
    const int q   = min(qg, NQ_ - 1);         // clamped for loads/compute
    const bool q_ok = qg < NQ_;

    __shared__ float sk[AK_][DH_];
    __shared__ float sv[AK_][DH_];

    const float scale = 0.17677669529663687f;   // 1/sqrt(32)
    float qv[DH_];
    const float* qp = qkv + ((long)(b * NQ_ + q)) * (3 * D_) + h * DH_;
#pragma unroll
    for (int d = 0; d < DH_; ++d) qv[d] = qp[d] * scale;

    float mmax = -1e30f, l = 0.0f;
    float acc[DH_];
#pragma unroll
    for (int d = 0; d < DH_; ++d) acc[d] = 0.0f;

    const float* kbase = qkv + (long)b * NQ_ * (3 * D_) + D_ + h * DH_;
    const float* vbase = kbase + D_;

    for (int j0 = 0; j0 < NQ_; j0 += AK_) {
        const int jn = min(AK_, NQ_ - j0);

        __syncthreads();   // previous chunk fully consumed
        // Cooperative stage: jn rows x 32 floats of K and V, float4-vectorized.
        for (int idx = tid; idx < jn * (DH_ / 4); idx += AQ_) {
            const int jr = idx >> 3;            // row within chunk
            const int dc = (idx & 7) << 2;      // float4 column
            const long  roff = (long)(j0 + jr) * (3 * D_) + dc;
            *(float4*)&sk[jr][dc] = *(const float4*)(kbase + roff);
            *(float4*)&sv[jr][dc] = *(const float4*)(vbase + roff);
        }
        // Prefetch next chunk while this one is processed (global_prefetch_b8).
        if (j0 + AK_ < NQ_) {
            __builtin_prefetch(kbase + (long)(j0 + AK_ + (tid >> 2)) * (3 * D_), 0, 1);
        }
        __syncthreads();

        for (int j = 0; j < jn; ++j) {
            float s = 0.0f;
#pragma unroll
            for (int d = 0; d < DH_; ++d) s += qv[d] * sk[j][d];
            const float nm   = fmaxf(mmax, s);
            const float cold = __expf(mmax - nm);
            const float p    = __expf(s - nm);
            l = l * cold + p;
#pragma unroll
            for (int d = 0; d < DH_; ++d) acc[d] = acc[d] * cold + p * sv[j][d];
            mmax = nm;
        }
    }

    if (q_ok) {
        const float inv = 1.0f / l;
        float* op = out + ((long)(b * NQ_ + qg)) * D_ + h * DH_;
#pragma unroll
        for (int d = 0; d < DH_; ++d) op[d] = acc[d] * inv;
    }
}

// ---------------------------------------------------------------------------
// Residual + LayerNorm over D=256. One block (256 threads = 8 waves) per row.
// ---------------------------------------------------------------------------
__global__ void ln_kernel(const float* __restrict__ x, const float* __restrict__ y,
                          const float* __restrict__ g, const float* __restrict__ bta,
                          float* __restrict__ out)
{
    const int  row  = blockIdx.x;
    const int  tid  = threadIdx.x;
    const int  wid  = tid >> 5;
    const int  lane = tid & 31;
    __shared__ float red[8];

    float v = x[(long)row * D_ + tid];
    if (y) v += y[(long)row * D_ + tid];

    float s = v;
#pragma unroll
    for (int o = 16; o; o >>= 1) s += __shfl_xor(s, o, 32);
    if (lane == 0) red[wid] = s;
    __syncthreads();
    float tot = 0.0f;
#pragma unroll
    for (int i = 0; i < 8; ++i) tot += red[i];
    const float mean = tot * (1.0f / D_);
    const float dv   = v - mean;
    __syncthreads();

    float s2 = dv * dv;
#pragma unroll
    for (int o = 16; o; o >>= 1) s2 += __shfl_xor(s2, o, 32);
    if (lane == 0) red[wid] = s2;
    __syncthreads();
    float tv = 0.0f;
#pragma unroll
    for (int i = 0; i < 8; ++i) tv += red[i];
    const float var = tv * (1.0f / D_);

    out[(long)row * D_ + tid] = dv * rsqrtf(var + 1e-5f) * g[tid] + bta[tid];
}

// ---------------------------------------------------------------------------
// Softmax over 16 attention weights per (b, q, h). In place on [57600,16].
// ---------------------------------------------------------------------------
__global__ void softmax16_kernel(float* __restrict__ aw)
{
    int t = blockIdx.x * blockDim.x + threadIdx.x;
    if (t >= NTOK_ * NH_) return;
    float* p = aw + (long)t * 16;
    float v[16], mx = -1e30f;
#pragma unroll
    for (int i = 0; i < 16; ++i) { v[i] = p[i]; mx = fmaxf(mx, v[i]); }
    float s = 0.0f;
#pragma unroll
    for (int i = 0; i < 16; ++i) { v[i] = __expf(v[i] - mx); s += v[i]; }
    const float inv = 1.0f / s;
#pragma unroll
    for (int i = 0; i < 16; ++i) p[i] = v[i] * inv;
}

// ---------------------------------------------------------------------------
// Multi-scale deformable attention sampling. One thread per (b, q, h).
// value rows are 128B-aligned -> float4 gathers lower to global_load_b128.
// value: [B, LV, 256] (f32, resident in 192MB L2); out: [B, NQ, 256].
// ---------------------------------------------------------------------------
__device__ __forceinline__ void gather_acc(float* acc, const float* __restrict__ vp, float wt)
{
#pragma unroll
    for (int i = 0; i < DH_ / 4; ++i) {
        const float4 v = *(const float4*)(vp + 4 * i);
        acc[4 * i + 0] += wt * v.x;
        acc[4 * i + 1] += wt * v.y;
        acc[4 * i + 2] += wt * v.z;
        acc[4 * i + 3] += wt * v.w;
    }
}

__global__ void deform_kernel(const float* __restrict__ value, const float* __restrict__ ref,
                              const float* __restrict__ off,   const float* __restrict__ aw,
                              const int* __restrict__ shapes,  const int* __restrict__ lstart,
                              float* __restrict__ out)
{
    int t = blockIdx.x * blockDim.x + threadIdx.x;
    if (t >= B_ * NQ_ * NH_) return;
    const int h = t & (NH_ - 1);
    const int q = (t >> 3) % NQ_;
    const int b = t / (NH_ * NQ_);

    float acc[DH_];
#pragma unroll
    for (int d = 0; d < DH_; ++d) acc[d] = 0.0f;

    const long  vbase = (long)b * LV_;
    const float* offr = off + ((long)(b * NQ_ + q)) * (NH_ * NL_ * NP_ * 2);
    const float* awr  = aw  + ((long)(b * NQ_ + q)) * (NH_ * NL_ * NP_);
    const float* refr = ref + ((long)(b * NQ_ + q)) * (NL_ * 2);

    for (int lvl = 0; lvl < NL_; ++lvl) {
        const int hl = shapes[2 * lvl];
        const int wl = shapes[2 * lvl + 1];
        const int st = lstart[lvl];
        const float rx = refr[2 * lvl];
        const float ry = refr[2 * lvl + 1];
        for (int p = 0; p < NP_; ++p) {
            const int oi = ((h * NL_ + lvl) * NP_ + p) * 2;
            // loc = ref + off / [w, h]
            const float gx = rx + offr[oi]     / (float)wl;
            const float gy = ry + offr[oi + 1] / (float)hl;
            const float xx = gx * (float)wl - 0.5f;
            const float yy = gy * (float)hl - 0.5f;
            const float fx = floorf(xx), fy = floorf(yy);
            const float lx = xx - fx,    ly = yy - fy;
            const int   x0 = (int)fx,    y0 = (int)fy;
            const float w  = awr[(h * NL_ + lvl) * NP_ + p];
            const float w00 = (1.0f - lx) * (1.0f - ly) * w;
            const float w01 = lx * (1.0f - ly) * w;
            const float w10 = (1.0f - lx) * ly * w;
            const float w11 = lx * ly * w;

            if (y0 >= 0 && y0 < hl && x0 >= 0 && x0 < wl)
                gather_acc(acc, value + (vbase + st + (long)y0 * wl + x0) * D_ + h * DH_, w00);
            if (y0 >= 0 && y0 < hl && x0 + 1 >= 0 && x0 + 1 < wl)
                gather_acc(acc, value + (vbase + st + (long)y0 * wl + x0 + 1) * D_ + h * DH_, w01);
            if (y0 + 1 >= 0 && y0 + 1 < hl && x0 >= 0 && x0 < wl)
                gather_acc(acc, value + (vbase + st + (long)(y0 + 1) * wl + x0) * D_ + h * DH_, w10);
            if (y0 + 1 >= 0 && y0 + 1 < hl && x0 + 1 >= 0 && x0 + 1 < wl)
                gather_acc(acc, value + (vbase + st + (long)(y0 + 1) * wl + x0 + 1) * D_ + h * DH_, w11);
        }
    }
    float* op = out + ((long)(b * NQ_ + q)) * D_ + h * DH_;
#pragma unroll
    for (int d = 0; d < DH_; ++d) op[d] = acc[d];
}

// ---------------------------------------------------------------------------
// Host-side orchestration
// ---------------------------------------------------------------------------
static inline void launch_gemm(const float* A, const float* W, const float* bias, float* C,
                               int M, int N, int K, int lda, int ldw, int ldc,
                               float alpha, int relu, hipStream_t stream)
{
    dim3 grid((N + 31) / 32, (M + 31) / 32, 1);
    gemm_wmma_kernel<<<grid, 32, 0, stream>>>(A, W, bias, C, M, N, K, lda, ldw, ldc, alpha, relu);
}

extern "C" void kernel_launch(void* const* d_in, const int* in_sizes, int n_in,
                              void* d_out, int out_size, void* d_ws, size_t ws_size,
                              hipStream_t stream)
{
    // ---- inputs (setup_inputs order) ----
    const float* query   = (const float*)d_in[0];   // [B,NQ,D]
    const float* refpts  = (const float*)d_in[1];   // [B,NQ,NL,2]
    const float* memory  = (const float*)d_in[2];   // [B,LV,D]
    const int*   shapes  = (const int*)  d_in[3];   // [NL,2]
    const int*   lstart  = (const int*)  d_in[4];   // [NL]
    // d_in[5]: memory_key_padding_mask -- identically false in this problem; no-op.
    const float* in_w    = (const float*)d_in[6];   // [768,256]
    const float* in_b    = (const float*)d_in[7];
    const float* out_w   = (const float*)d_in[8];   // [256,256]
    const float* out_b   = (const float*)d_in[9];
    const float* ln1_g   = (const float*)d_in[10];
    const float* ln1_b   = (const float*)d_in[11];
    const float* val_w   = (const float*)d_in[12];  // [256,256]
    const float* val_b   = (const float*)d_in[13];
    const float* so_w    = (const float*)d_in[14];  // [256,256]
    const float* so_b    = (const float*)d_in[15];
    const float* awp_w   = (const float*)d_in[16];  // [128,256]
    const float* awp_b   = (const float*)d_in[17];
    const float* ca_w    = (const float*)d_in[18];  // [256,256]
    const float* ca_b    = (const float*)d_in[19];
    const float* ln2_g   = (const float*)d_in[20];
    const float* ln2_b   = (const float*)d_in[21];
    const float* f1_w    = (const float*)d_in[22];  // [1024,256]
    const float* f1_b    = (const float*)d_in[23];
    const float* f2_w    = (const float*)d_in[24];  // [256,1024]
    const float* f2_b    = (const float*)d_in[25];
    const float* ln3_g   = (const float*)d_in[26];
    const float* ln3_b   = (const float*)d_in[27];
    float* outp = (float*)d_out;

    // ---- workspace layout (floats) ----
    float* ws = (float*)d_ws;
    size_t o = 0;
    float* qkv   = ws + o; o += (size_t)NTOK_ * 3 * D_;   // 7200 x 768
    float* sa_o  = ws + o; o += (size_t)NTOK_ * D_;       // attention output
    float* sa_p  = ws + o; o += (size_t)NTOK_ * D_;       // out-proj result
    float* q1    = ws + o; o += (size_t)NTOK_ * D_;       // after LN1
    float* value = ws + o; o += (size_t)MTOK_ * D_;       // projected memory
    float* offb  = ws + o; o += (size_t)NTOK_ * (NH_*NL_*NP_*2);
    float* awb   = ws + o; o += (size_t)NTOK_ * (NH_*NL_*NP_);
    float* msb   = ws + o; o += (size_t)NTOK_ * D_;       // deformable attn out
    float* cab   = ws + o; o += (size_t)NTOK_ * D_;       // ca_out proj
    float* q2    = ws + o; o += (size_t)NTOK_ * D_;       // after LN2
    float* ffnh  = ws + o; o += (size_t)NTOK_ * DFF_;     // relu hidden
    float* ffno  = ws + o; o += (size_t)NTOK_ * D_;       // ffn output
    (void)ws_size; (void)in_sizes; (void)n_in; (void)out_size;

    const int nbhq = B_ * NQ_ * NH_;  // 57600

    // 1) qkv projection (WMMA)
    launch_gemm(query, in_w, in_b, qkv, NTOK_, 3 * D_, D_, D_, D_, 3 * D_, 1.0f, 0, stream);

    // 2) fused self-attention (LDS-tiled K/V)
    {
        dim3 grid((NQ_ + AQ_ - 1) / AQ_, NH_, B_);
        self_attn_kernel<<<grid, AQ_, 0, stream>>>(qkv, sa_o);
    }

    // 3) out projection + residual LN1
    launch_gemm(sa_o, out_w, out_b, sa_p, NTOK_, D_, D_, D_, D_, D_, 1.0f, 0, stream);
    ln_kernel<<<NTOK_, 256, 0, stream>>>(query, sa_p, ln1_g, ln1_b, q1);

    // 4) value projection over all memory tokens (WMMA)
    launch_gemm(memory, val_w, val_b, value, MTOK_, D_, D_, D_, D_, D_, 1.0f, 0, stream);

    // 5) sampling offsets + attention weights (+softmax over 16)
    launch_gemm(q1, so_w, so_b, offb, NTOK_, NH_*NL_*NP_*2, D_, D_, D_, NH_*NL_*NP_*2, 1.0f, 0, stream);
    launch_gemm(q1, awp_w, awp_b, awb, NTOK_, NH_*NL_*NP_,   D_, D_, D_, NH_*NL_*NP_,   1.0f, 0, stream);
    softmax16_kernel<<<(nbhq + 255) / 256, 256, 0, stream>>>(awb);

    // 6) deformable bilinear sampling
    deform_kernel<<<(nbhq + 255) / 256, 256, 0, stream>>>(value, refpts, offb, awb,
                                                          shapes, lstart, msb);

    // 7) ca_out projection + residual LN2
    launch_gemm(msb, ca_w, ca_b, cab, NTOK_, D_, D_, D_, D_, D_, 1.0f, 0, stream);
    ln_kernel<<<NTOK_, 256, 0, stream>>>(q1, cab, ln2_g, ln2_b, q2);

    // 8) FFN (relu fused into GEMM1 epilogue) + residual LN3 -> d_out
    launch_gemm(q2, f1_w, f1_b, ffnh, NTOK_, DFF_, D_, D_, D_, DFF_, 1.0f, 1, stream);
    launch_gemm(ffnh, f2_w, f2_b, ffno, NTOK_, D_, DFF_, DFF_, DFF_, D_, 1.0f, 0, stream);
    ln_kernel<<<NTOK_, 256, 0, stream>>>(q2, ffno, ln3_g, ln3_b, outp);
}
